// TransformerMusic_27023934227128
// MI455X (gfx1250) — matrix operand order repaired and checked by reading, not compiled, not run on hardware
//
#include <hip/hip_runtime.h>
#include <math.h>

// ---------------- model constants ----------------
#define D_IN   4
#define HIST   100
#define KSZ    4
#define STRD   2
#define CCH    128
#define NHEAD  8
#define DH     16
#define NL     4
#define HFF    512
#define SEQL   108
#define LENC   49          // (100-4)/2+1
#define STEPS  8           // SEQ - HIST
#define OUTDIM 258
#define BATCH  256
#define MROWS  (BATCH*LENC)   // 12544, multiple of 32

typedef __attribute__((ext_vector_type(16))) _Float16 v16h;
typedef __attribute__((ext_vector_type(8)))  float    v8f;
typedef __attribute__((ext_vector_type(4))) unsigned int u32x4;
typedef __attribute__((ext_vector_type(8))) int          i32x8;
typedef __attribute__((ext_vector_type(4))) int          i32x4;
typedef _Float16 half_t;

// ---------------------------------------------------------------------
// TDM issue: 2-D tile 32(K) x 64(N) f16 from row-major (N,K) weight into
// LDS at byte offset ldsOff, padded to a 40-half (80 B) row stride.
// OOB rows/cols (dims measured from tile start) are hardware zero-filled.
// ---------------------------------------------------------------------
__device__ __forceinline__ void tdm_issue_w(const half_t* W, int N, int K,
                                            int blockN, int k0, unsigned ldsOff)
{
    const unsigned long long gaddr =
        (unsigned long long)(uintptr_t)W + 2ull * ((size_t)blockN * K + k0);
    u32x4 g0 = {};
    g0[0] = 1u;                                   // count = 1 (valid D#)
    g0[1] = ldsOff;                               // lds_addr (bytes)
    g0[2] = (unsigned)(gaddr & 0xFFFFFFFFu);      // global_addr[31:0]
    g0[3] = (unsigned)((gaddr >> 32) & 0x01FFFFFFu) | 0x80000000u; // [56:32] | type=2
    i32x8 g1 = {};
    // data_size=1(2B) | pad_enable | pad_interval=3(16 DW=64B) | pad_amount=3(4 DW=16B)
    g1[0] = (1 << 16) | (1 << 20) | (3 << 22) | (3 << 25);
    const unsigned td0 = (unsigned)(K - k0);      // remaining x extent
    const unsigned td1 = (unsigned)(N - blockN);  // remaining y extent
    g1[1] = (int)(td0 << 16);                     // tensor_dim0 lo16
    g1[2] = (int)((td0 >> 16) | (td1 << 16));     // dim0 hi16 | dim1 lo16
    g1[3] = (int)((td1 >> 16) | (32u << 16));     // dim1 hi16 | tile_dim0=32
    g1[4] = 64;                                   // tile_dim1=64, tile_dim2=0
    g1[5] = K;                                    // tensor_dim0_stride[31:0]
    g1[6] = 0;
    g1[7] = 0;
    i32x4 z4 = {};
    i32x8 z8 = {};
    __builtin_amdgcn_tensor_load_to_lds(g0, g1, z4, z4, z8, 0);
}

// =====================================================================
// Generic fused GEMM:  out[M,N] = act( A[M,K](f16) * W[N,K]^T(f16) + bias + residual )
// Block = 256 threads = 8 waves; block tile 32(M) x 64(N); K step 32.
// Double-buffered pipeline: TDM (W tile) + global->reg (A tile) for chunk
// c+1 are issued before computing chunk c; tensorcnt wait + one barrier
// per iteration publish the next buffer, hiding the fill latency behind
// ds_load_b128 + v_wmma_f32_16x16x32_f16.
// Requires: M % 32 == 0, K % 32 == 0 (true for all call sites).
// act: 0=none, 1=relu, 2=gelu(exact erf)
// =====================================================================
__global__ __launch_bounds__(256)
void gemm_wmma(const half_t* __restrict__ A, const half_t* __restrict__ W,
               const float* __restrict__ bias, const float* __restrict__ residual,
               float* __restrict__ C32, half_t* __restrict__ C16,
               int M, int N, int K, int act)
{
    __shared__ half_t As[2][32][40];   // row stride 80 B (multiple of 16 B)
    __shared__ half_t Ws[2][64][40];

    const int tid   = threadIdx.x;
    const int lane  = tid & 31;
    const int wave  = tid >> 5;
    const int waveM = wave >> 2;     // 0..1
    const int waveN = wave & 3;      // 0..3
    const int grp   = lane >> 4;     // lane group 0/1 (ISA fragment layout)
    const int l16   = lane & 15;
    const int blockM = blockIdx.y * 32;
    const int blockN = blockIdx.x * 64;

    const unsigned ldsW0 = (unsigned)(unsigned long long)(uintptr_t)&Ws[0][0][0];
    const unsigned ldsW1 = (unsigned)(unsigned long long)(uintptr_t)&Ws[1][0][0];

    const int nchunks = K >> 5;
    const int r  = tid >> 3;          // 0..31 (A-tile row)
    const int cg = tid & 7;           // 4 cols per thread
    const half_t* Arow = A + (size_t)(blockM + r) * K + cg * 4;

    v8f acc = {};

    // ---- prologue: fill buffer 0 ----
    uint2 av = *(const uint2*)(Arow);                       // chunk 0
    if (wave == 0) tdm_issue_w(W, N, K, blockN, 0, ldsW0);
    *(uint2*)&As[0][r][cg * 4] = av;
    if (wave == 0) __builtin_amdgcn_s_wait_tensorcnt(0);
    __syncthreads();

    for (int c = 0; c < nchunks; ++c) {
        const int cur = c & 1, nxt = cur ^ 1;
        if (c + 1 < nchunks) {
            const int k1 = (c + 1) << 5;
            av = *(const uint2*)(Arow + k1);                // A chunk c+1 -> regs
            if (wave == 0)
                tdm_issue_w(W, N, K, blockN, k1, nxt ? ldsW1 : ldsW0);
            if (c + 2 < nchunks)                            // global_prefetch_b8
                __builtin_prefetch(Arow + ((c + 2) << 5), 0, 1);
        }

        // ---- fragments in contiguous k order k = 16*grp + e for A and B (-> ds_load_b128) ----
        v16h a, b;
        #pragma unroll
        for (int e = 0; e < 16; ++e) {           // A 16x32: lane=row, element->K
            const int v = e >> 1, h = e & 1;
            const int kk = 16 * grp + e;
            a[e] = As[cur][waveM * 16 + l16][kk];
        }
        #pragma unroll
        for (int e = 0; e < 16; ++e)             // B 32x16: lane=col, element->K
            b[e] = Ws[cur][waveN * 16 + l16][16 * grp + e];

        acc = __builtin_amdgcn_wmma_f32_16x16x32_f16(
                  false, a, false, b, (short)0, acc, false, false);

        if (c + 1 < nchunks) {
            *(uint2*)&As[nxt][r][cg * 4] = av;              // publish A chunk c+1
            if (wave == 0) __builtin_amdgcn_s_wait_tensorcnt(0);
            __syncthreads();                                // publish buffer nxt
        }
    }

    // ---- epilogue: bias + residual + activation, dual-precision store ----
    const int col = blockN + waveN * 16 + l16;
    if (col < N) {
        const float bv = bias ? bias[col] : 0.0f;
        #pragma unroll
        for (int rr = 0; rr < 8; ++rr) {         // C/D: VGPR rr -> row rr+8*grp
            const int row = blockM + waveM * 16 + rr + 8 * grp;
            float val = acc[rr] + bv;
            if (residual) val += residual[(size_t)row * N + col];
            if (act == 1)      val = fmaxf(val, 0.0f);
            else if (act == 2) val = 0.5f * val * (1.0f + erff(val * 0.70710678118654752f));
            if (C32) C32[(size_t)row * N + col] = val;
            if (C16) C16[(size_t)row * N + col] = (half_t)val;
        }
    }
}

// =====================================================================
// Fused per-(batch, head) attention.  seq=49, dh=16.  qkv is [M,384] f32.
// =====================================================================
__global__ __launch_bounds__(64)
void attn_kernel(const float* __restrict__ qkv, half_t* __restrict__ o16)
{
    __shared__ float sq[LENC][DH], sk[LENC][DH], sv[LENC][DH];
    const int bh = blockIdx.x;
    const int b = bh / NHEAD, h = bh % NHEAD;
    const int tid = threadIdx.x;

    for (int idx = tid; idx < LENC * DH; idx += 64) {
        const int t = idx / DH, d = idx % DH;
        const size_t base = (size_t)(b * LENC + t) * (3 * CCH) + h * DH + d;
        sq[t][d] = qkv[base];
        sk[t][d] = qkv[base + CCH];
        sv[t][d] = qkv[base + 2 * CCH];
    }
    __syncthreads();

    const int t = tid;
    if (t < LENC) {
        float mx = -3.0e38f;
        for (int j = 0; j < LENC; ++j) {
            float s = 0.0f;
            #pragma unroll
            for (int d = 0; d < DH; ++d) s += sq[t][d] * sk[j][d];
            mx = fmaxf(mx, s * 0.25f);          // 1/sqrt(16)
        }
        float sum = 0.0f, accv[DH];
        #pragma unroll
        for (int d = 0; d < DH; ++d) accv[d] = 0.0f;
        for (int j = 0; j < LENC; ++j) {
            float s = 0.0f;
            #pragma unroll
            for (int d = 0; d < DH; ++d) s += sq[t][d] * sk[j][d];
            const float w = expf(s * 0.25f - mx);
            sum += w;
            #pragma unroll
            for (int d = 0; d < DH; ++d) accv[d] += w * sv[j][d];
        }
        const float inv = 1.0f / sum;
        const size_t orow = (size_t)(b * LENC + t) * CCH + h * DH;
        #pragma unroll
        for (int d = 0; d < DH; ++d) o16[orow + d] = (half_t)(accv[d] * inv);
    }
}

// =====================================================================
// LayerNorm: one wave per 128-wide row (wave32: 4 cols/lane), shfl reduce.
// =====================================================================
__global__ __launch_bounds__(256)
void ln_kernel(const float* __restrict__ X, const float* __restrict__ gamma,
               const float* __restrict__ beta,
               float* __restrict__ Y32, half_t* __restrict__ Y16, int M)
{
    const int wave = threadIdx.x >> 5;
    const int lane = threadIdx.x & 31;
    const int row = blockIdx.x * 8 + wave;
    if (row >= M) return;
    const float* x = X + (size_t)row * CCH;
    float vv[4];
    float s1 = 0.0f, s2 = 0.0f;
    #pragma unroll
    for (int i = 0; i < 4; ++i) {
        vv[i] = x[lane + 32 * i];
        s1 += vv[i]; s2 += vv[i] * vv[i];
    }
    #pragma unroll
    for (int m = 16; m > 0; m >>= 1) {
        s1 += __shfl_xor(s1, m, 32);
        s2 += __shfl_xor(s2, m, 32);
    }
    const float mean = s1 * (1.0f / CCH);
    const float var  = s2 * (1.0f / CCH) - mean * mean;   // population var (jnp.var)
    const float rs   = rsqrtf(var + 1e-5f);
    #pragma unroll
    for (int i = 0; i < 4; ++i) {
        const int c = lane + 32 * i;
        const float val = (vv[i] - mean) * rs * gamma[c] + beta[c];
        Y32[(size_t)row * CCH + c] = val;
        Y16[(size_t)row * CCH + c] = (half_t)val;
    }
}

// =====================================================================
// Conv1d (D=4 -> C=128, k=4, stride=2).  One block per (b, l); thread = out ch.
// =====================================================================
__global__ __launch_bounds__(128)
void conv_kernel(const float* __restrict__ hist, const float* __restrict__ cw,
                 const float* __restrict__ cb,
                 float* __restrict__ z32, half_t* __restrict__ z16)
{
    __shared__ float sh[16];
    const int bl = blockIdx.x;               // b*LENC + l
    const int b = bl / LENC, l = bl % LENC;
    const int c = threadIdx.x;
    if (c < 16) {
        const int d = c >> 2, k = c & 3;
        sh[c] = hist[(size_t)b * (D_IN * HIST) + d * HIST + l * STRD + k];
    }
    __syncthreads();
    float acc = cb[c];
    #pragma unroll
    for (int i = 0; i < 16; ++i) acc += sh[i] * cw[c * 16 + i];
    const size_t o = (size_t)bl * CCH + c;
    z32[o] = acc;
    z16[o] = (half_t)acc;
}

// =====================================================================
// hist0[b, d, t] = x[b, t, d]   (t < 100)
// =====================================================================
__global__ void init_hist(const float* __restrict__ x, float* __restrict__ hist)
{
    const int b = blockIdx.x;
    for (int idx = threadIdx.x; idx < D_IN * HIST; idx += blockDim.x) {
        const int d = idx / HIST, t = idx % HIST;
        hist[(size_t)b * (D_IN * HIST) + idx] = x[(size_t)b * SEQL * D_IN + t * D_IN + d];
    }
}

__global__ void f32_to_f16(const float* __restrict__ s, half_t* __restrict__ d, int n)
{
    const int i = blockIdx.x * blockDim.x + threadIdx.x;
    if (i < n) d[i] = (half_t)s[i];
}

// =====================================================================
// JAX threefry2x32 machinery (exact re-implementation of jax.random bits)
// =====================================================================
__device__ __forceinline__ unsigned rotl32(unsigned v, int r) {
    return (v << r) | (v >> (32 - r));
}
__device__ __forceinline__ void tf4(unsigned& x0, unsigned& x1, int a, int b, int c, int d)
{
    x0 += x1; x1 = rotl32(x1, a); x1 ^= x0;
    x0 += x1; x1 = rotl32(x1, b); x1 ^= x0;
    x0 += x1; x1 = rotl32(x1, c); x1 ^= x0;
    x0 += x1; x1 = rotl32(x1, d); x1 ^= x0;
}
__device__ __forceinline__ uint2 threefry2x32(unsigned k0, unsigned k1, unsigned x0, unsigned x1)
{
    const unsigned ks2 = k0 ^ k1 ^ 0x1BD11BDAu;
    x0 += k0; x1 += k1;
    tf4(x0, x1, 13, 15, 26, 6);  x0 += k1;  x1 += ks2 + 1u;
    tf4(x0, x1, 17, 29, 16, 24); x0 += ks2; x1 += k0 + 2u;
    tf4(x0, x1, 13, 15, 26, 6);  x0 += k0;  x1 += k1 + 3u;
    tf4(x0, x1, 17, 29, 16, 24); x0 += k1;  x1 += ks2 + 4u;
    tf4(x0, x1, 13, 15, 26, 6);  x0 += ks2; x1 += k0 + 5u;
    return make_uint2(x0, x1);
}
// element m of jax random_bits(key, 32, (n,)) with n even:
__device__ __forceinline__ unsigned jax_bits(unsigned k0, unsigned k1, unsigned m, unsigned n)
{
    const unsigned h = n >> 1;
    if (m < h) return threefry2x32(k0, k1, m, m + h).x;
    else       return threefry2x32(k0, k1, m - h, m).y;
}
__device__ __forceinline__ float bits_to_unit(unsigned b)   // uniform [0,1)
{
    return __uint_as_float((b >> 9) | 0x3F800000u) - 1.0f;
}

// =====================================================================
// Sampling + history update.  One block (128 thr) per batch element.
// step_keys = split(key(42), 32).reshape(8,4) computed on the fly.
// =====================================================================
__global__ __launch_bounds__(128)
void sample_kernel(const float* __restrict__ out258, const float* __restrict__ histIn,
                   float* __restrict__ histOut, float* __restrict__ notes_out, int s)
{
    __shared__ float sval[128];
    __shared__ int   sidx[128];
    __shared__ float sh_note[4];

    const int b = blockIdx.x;
    const int tid = threadIdx.x;

    unsigned kk0[4], kk1[4];                 // keys k1..k4 for this step
    #pragma unroll
    for (int j = 0; j < 4; ++j) {
        const unsigned i = (unsigned)(s * 4 + j);
        kk0[j] = jax_bits(0u, 42u, 2u * i,      64u);
        kk1[j] = jax_bits(0u, 42u, 2u * i + 1u, 64u);
    }
    const float* o = out258 + (size_t)b * OUTDIM;

    int cat_idx[2];
    #pragma unroll
    for (int cat = 0; cat < 2; ++cat) {      // cat 0 -> pitch(k3), 1 -> vel(k4)
        const unsigned m = (unsigned)(b * 128 + tid);
        float u = bits_to_unit(jax_bits(kk0[2 + cat], kk1[2 + cat], m, 32768u));
        const float tiny = 1.1754943508e-38f;
        u = u * (1.0f - tiny) + tiny;        // jax gumbel uses minval=tiny
        const float g = -logf(-logf(u));
        sval[tid] = o[(cat == 0 ? 2 : 130) + tid] + g;
        sidx[tid] = tid;
        __syncthreads();
        for (int ofs = 64; ofs > 0; ofs >>= 1) {
            if (tid < ofs) {
                const float v2 = sval[tid + ofs]; const int i2 = sidx[tid + ofs];
                if (v2 > sval[tid] || (v2 == sval[tid] && i2 < sidx[tid])) {
                    sval[tid] = v2; sidx[tid] = i2;   // first-max tie-break
                }
            }
            __syncthreads();
        }
        cat_idx[cat] = sidx[0];
        __syncthreads();
    }

    if (tid == 0) {
        const float last_start = histIn[(size_t)b * (D_IN * HIST) + (HIST - 1)];
        const float u1 = bits_to_unit(jax_bits(kk0[0], kk1[0], (unsigned)b, 256u));
        const float start = -log1pf(-u1) * expf(-o[0]) + last_start;
        const float u2 = bits_to_unit(jax_bits(kk0[1], kk1[1], (unsigned)b, 256u));
        const float end = -log1pf(-u2) * expf(-o[1]) + start;
        const float pitch = (float)cat_idx[0] * (1.0f / 16.0f);
        const float vel   = (float)cat_idx[1] * (1.0f / 16.0f);
        sh_note[0] = start; sh_note[1] = end; sh_note[2] = pitch; sh_note[3] = vel;
        float* no = notes_out + ((size_t)b * STEPS + s) * 4;
        no[0] = start; no[1] = end; no[2] = pitch; no[3] = vel;
    }
    __syncthreads();

    for (int idx = tid; idx < D_IN * HIST; idx += 128) {
        const int d = idx / HIST, t = idx % HIST;
        const size_t base = (size_t)b * (D_IN * HIST) + d * HIST;
        histOut[base + t] = (t < HIST - 1) ? histIn[base + t + 1] : sh_note[d];
    }
}

// =====================================================================
// host-side orchestration
// =====================================================================
extern "C" void kernel_launch(void* const* d_in, const int* in_sizes, int n_in,
                              void* d_out, int out_size, void* d_ws, size_t ws_size,
                              hipStream_t stream)
{
    const float* x      = (const float*)d_in[0];
    const float* conv_w = (const float*)d_in[1];
    const float* conv_b = (const float*)d_in[2];
    const float* qkv_w  = (const float*)d_in[3];
    const float* qkv_b  = (const float*)d_in[4];
    const float* aow    = (const float*)d_in[5];
    const float* aob    = (const float*)d_in[6];
    const float* ln1w   = (const float*)d_in[7];
    const float* ln1b   = (const float*)d_in[8];
    const float* ln2w   = (const float*)d_in[9];
    const float* ln2b   = (const float*)d_in[10];
    const float* ff1w   = (const float*)d_in[11];
    const float* ff1b   = (const float*)d_in[12];
    const float* ff2w   = (const float*)d_in[13];
    const float* ff2b   = (const float*)d_in[14];
    const float* dec1w  = (const float*)d_in[15];
    const float* dec1b  = (const float*)d_in[16];
    const float* dec2w  = (const float*)d_in[17];
    const float* dec2b  = (const float*)d_in[18];
    const float* dec3w  = (const float*)d_in[19];
    const float* dec3b  = (const float*)d_in[20];
    float* out = (float*)d_out;

    // ---- workspace carve-up (~56 MB) ----
    char* base = (char*)d_ws;
    size_t off = 0;
    auto take = [&](size_t bytes) -> char* {
        char* r = base + off;
        off = (off + bytes + 255) & ~(size_t)255;
        return r;
    };
    float*  histA  = (float*)take((size_t)BATCH * D_IN * HIST * 4);
    float*  histB  = (float*)take((size_t)BATCH * D_IN * HIST * 4);
    float*  z32    = (float*)take((size_t)MROWS * CCH * 4);
    half_t* z16    = (half_t*)take((size_t)MROWS * CCH * 2);
    float*  qkv32  = (float*)take((size_t)MROWS * 3 * CCH * 4);   // 19.3 MB
    half_t* g16    = (half_t*)qkv32;                              // alias: ff1 out, qkv dead by then
    half_t* o16    = (half_t*)take((size_t)MROWS * CCH * 2);
    float*  y32    = (float*)take((size_t)MROWS * CCH * 4);
    half_t* d1_16  = (half_t*)take((size_t)BATCH * 1024 * 2);
    half_t* d2_16  = (half_t*)take((size_t)BATCH * 512 * 2);
    float*  out258 = (float*)take((size_t)BATCH * OUTDIM * 4);
    half_t* qkvW16 = (half_t*)take((size_t)NL * 3 * CCH * CCH * 2);
    half_t* aoW16  = (half_t*)take((size_t)NL * CCH * CCH * 2);
    half_t* ff1W16 = (half_t*)take((size_t)NL * HFF * CCH * 2);
    half_t* ff2W16 = (half_t*)take((size_t)NL * CCH * HFF * 2);
    half_t* dec1W16= (half_t*)take((size_t)1024 * (LENC * CCH) * 2);
    half_t* dec2W16= (half_t*)take((size_t)512 * 1024 * 2);
    half_t* dec3W16= (half_t*)take((size_t)OUTDIM * HFF * 2);
    (void)ws_size; (void)in_sizes; (void)n_in; (void)out_size;

    // ---- one-time (per launch) weight conversion to f16 ----
    auto cvt = [&](const float* s, half_t* d, int n) {
        f32_to_f16<<<(n + 255) / 256, 256, 0, stream>>>(s, d, n);
    };
    cvt(qkv_w,  qkvW16,  NL * 3 * CCH * CCH);
    cvt(aow,    aoW16,   NL * CCH * CCH);
    cvt(ff1w,   ff1W16,  NL * HFF * CCH);
    cvt(ff2w,   ff2W16,  NL * CCH * HFF);
    cvt(dec1w,  dec1W16, 1024 * LENC * CCH);
    cvt(dec2w,  dec2W16, 512 * 1024);
    cvt(dec3w,  dec3W16, OUTDIM * HFF);

    init_hist<<<BATCH, 128, 0, stream>>>(x, histA);

    for (int s = 0; s < STEPS; ++s) {
        float* hin  = (s & 1) ? histB : histA;
        float* hout = (s & 1) ? histA : histB;

        conv_kernel<<<BATCH * LENC, 128, 0, stream>>>(hin, conv_w, conv_b, z32, z16);

        for (int i = 0; i < NL; ++i) {
            // QKV: [12544,128] x [384,128]^T
            gemm_wmma<<<dim3(6, MROWS / 32), 256, 0, stream>>>(
                z16, qkvW16 + (size_t)i * 3 * CCH * CCH, qkv_b + i * 3 * CCH,
                nullptr, qkv32, nullptr, MROWS, 3 * CCH, CCH, 0);

            attn_kernel<<<BATCH * NHEAD, 64, 0, stream>>>(qkv32, o16);

            // out-proj + residual(z): y = o @ Wout^T + b + z
            gemm_wmma<<<dim3(2, MROWS / 32), 256, 0, stream>>>(
                o16, aoW16 + (size_t)i * CCH * CCH, aob + i * CCH,
                z32, y32, nullptr, MROWS, CCH, CCH, 0);

            ln_kernel<<<MROWS / 8, 256, 0, stream>>>(
                y32, ln1w + i * CCH, ln1b + i * CCH, z32, z16, MROWS);

            // FF1 + exact GELU -> f16
            gemm_wmma<<<dim3(8, MROWS / 32), 256, 0, stream>>>(
                z16, ff1W16 + (size_t)i * HFF * CCH, ff1b + i * HFF,
                nullptr, nullptr, g16, MROWS, HFF, CCH, 2);

            // FF2 + residual(ln1 out)
            gemm_wmma<<<dim3(2, MROWS / 32), 256, 0, stream>>>(
                g16, ff2W16 + (size_t)i * CCH * HFF, ff2b + i * CCH,
                z32, y32, nullptr, MROWS, CCH, HFF, 0);

            ln_kernel<<<MROWS / 8, 256, 0, stream>>>(
                y32, ln2w + i * CCH, ln2b + i * CCH, z32, z16, MROWS);
        }

        // decoder: z16 viewed as [256, 6272] (row-major contiguity matches)
        gemm_wmma<<<dim3(16, BATCH / 32), 256, 0, stream>>>(
            z16, dec1W16, dec1b, nullptr, nullptr, d1_16, BATCH, 1024, LENC * CCH, 1);
        gemm_wmma<<<dim3(8, BATCH / 32), 256, 0, stream>>>(
            d1_16, dec2W16, dec2b, nullptr, nullptr, d2_16, BATCH, 512, 1024, 1);
        gemm_wmma<<<dim3(5, BATCH / 32), 256, 0, stream>>>(
            d2_16, dec3W16, dec3b, nullptr, out258, nullptr, BATCH, OUTDIM, 512, 0);

        sample_kernel<<<BATCH, 128, 0, stream>>>(out258, hin, hout, out, s);
    }
}
